// FeatureAggregation_39548058862067
// MI455X (gfx1250) — matrix operand order repaired and checked by reading, not compiled, not run on hardware
//
#include <hip/hip_runtime.h>

typedef _Float16 f16;
typedef __attribute__((ext_vector_type(16))) _Float16 v16h;
typedef __attribute__((ext_vector_type(8)))  _Float16 v8h;
typedef __attribute__((ext_vector_type(8)))  float    v8f;

#define BATCH 8
#define NPTS  16384
#define KNB   16      // stored neighbor dim
#define CF    64      // feature channels
#define CIN   68      // 64 + 3 diff + 1 dist
#define CPAD  96      // padded to multiple of 32 for the K loop
#define L1    128
#define L2    64
#define L3    64
#define WAVES 4
#define PTS_PER_BLK 64        // WAVES * 16
#define THREADS 128

union AU { v16h v; v8h h[2]; };

__global__ __launch_bounds__(THREADS)
void fa_fused_wmma(const float* __restrict__ src_xyz,
                   const float* __restrict__ tgt_xyz,
                   const float* __restrict__ feature,
                   const float* __restrict__ W1, const float* __restrict__ g1,
                   const float* __restrict__ b1, const float* __restrict__ m1, const float* __restrict__ v1,
                   const float* __restrict__ W2, const float* __restrict__ g2,
                   const float* __restrict__ b2, const float* __restrict__ m2, const float* __restrict__ v2,
                   const float* __restrict__ W3, const float* __restrict__ g3,
                   const float* __restrict__ b3, const float* __restrict__ m3, const float* __restrict__ v3,
                   float* __restrict__ out)
{
    // ---- LDS (≈134 KB -> 2 blocks / WGP) ----
    __shared__ __attribute__((aligned(32))) f16 sW1[L1 * CPAD];            // 24 KB
    __shared__ __attribute__((aligned(32))) f16 sW2[L2 * L1];              // 16 KB
    __shared__ __attribute__((aligned(32))) f16 sW3[L3 * L2];              //  8 KB
    __shared__ __attribute__((aligned(32))) float sInv1[L1], sBeta1[L1];
    __shared__ __attribute__((aligned(32))) float sInv2[L2], sBeta2[L2];
    __shared__ __attribute__((aligned(32))) float sInv3[L3], sBeta3[L3];
    __shared__ __attribute__((aligned(32))) f16 sX[WAVES * 3 * 16 * CPAD]; // 36 KB  [wave][k][col][ch]
    __shared__ __attribute__((aligned(32))) f16 sScr[WAVES * 3 * 16 * L1]; // 48 KB  [wave][k][col][ch]

    const int t  = threadIdx.x;
    const int p0 = blockIdx.x * PTS_PER_BLK;

    // ---- stage weights (f32 -> f16, W1 zero-padded to CPAD cols) ----
    for (int i = t; i < L1 * CPAD; i += THREADS) {
        int r = i / CPAD, c = i - r * CPAD;
        sW1[i] = (f16)((c < CIN) ? W1[r * CIN + c] : 0.0f);
    }
    for (int i = t; i < L2 * L1; i += THREADS) sW2[i] = (f16)W2[i];
    for (int i = t; i < L3 * L2; i += THREADS) sW3[i] = (f16)W3[i];
    if (t < L1) { float iv = g1[t] / sqrtf(v1[t] + 1e-5f); sInv1[t] = iv; sBeta1[t] = b1[t] - m1[t] * iv; }
    if (t < L2) { float iv = g2[t] / sqrtf(v2[t] + 1e-5f); sInv2[t] = iv; sBeta2[t] = b2[t] - m2[t] * iv; }
    if (t < L3) { float iv = g3[t] / sqrtf(v3[t] + 1e-5f); sInv3[t] = iv; sBeta3[t] = b3[t] - m3[t] * iv; }

    // ---- stage X: feature channels (coalesced across n; k<3 of each 64B line) ----
    for (int i = t; i < CF * PTS_PER_BLK; i += THREADS) {
        int c = i >> 6, pc = i & 63;
        int q = p0 + pc, bb = q >> 14, n = q & (NPTS - 1);
        const float* fp = feature + ((((size_t)bb * CF + c) * NPTS + n) * KNB);
        int w = pc >> 4, ln = pc & 15;
        sX[((w * 3 + 0) * 16 + ln) * CPAD + c] = (f16)fp[0];
        sX[((w * 3 + 1) * 16 + ln) * CPAD + c] = (f16)fp[1];
        sX[((w * 3 + 2) * 16 + ln) * CPAD + c] = (f16)fp[2];
    }
    // ---- stage X: diff (3) + dist (1) + zero pad ----
    for (int i = t; i < 3 * PTS_PER_BLK; i += THREADS) {
        int k = i >> 6, pc = i & 63;
        int q = p0 + pc, bb = q >> 14, n = q & (NPTS - 1);
        float dist = 0.0f, dx[3];
#pragma unroll
        for (int c3 = 0; c3 < 3; ++c3) {
            float s  = src_xyz[(((size_t)bb * 3 + c3) * NPTS + n) * KNB + k];
            float tg = tgt_xyz[((size_t)bb * 3 + c3) * NPTS + n];
            float d = s - tg; dx[c3] = d; dist += d * d;
        }
        int w = pc >> 4, ln = pc & 15;
        f16* col = &sX[((w * 3 + k) * 16 + ln) * CPAD];
        col[64] = (f16)dx[0]; col[65] = (f16)dx[1]; col[66] = (f16)dx[2]; col[67] = (f16)dist;
#pragma unroll
        for (int c = CIN; c < CPAD; ++c) col[c] = (f16)0.0f;
    }
    __syncthreads();

    // ---- per-wave fused MLP: all 3 neighbors as parallel WMMA streams ----
    const int lane = t & 31;
    const int wv   = t >> 5;
    const int ln   = lane & 15;
    const int hl   = lane >> 4;

    // B fragments of X: [neighbor][kc], per lane 16 contiguous K of column ln
    v16h bx[3][3];
#pragma unroll
    for (int k = 0; k < 3; ++k) {
        const f16* xcol = &sX[((wv * 3 + k) * 16 + ln) * CPAD];
#pragma unroll
        for (int kc = 0; kc < 3; ++kc)
            bx[k][kc] = *(const v16h*)(xcol + kc * 32 + hl * 16);
    }

    // ---- layer 1: (128 x 96) @ (96 x 16) x3 neighbors ----
#pragma unroll
    for (int mt = 0; mt < 8; ++mt) {
        v8f acc[3];
#pragma unroll
        for (int k = 0; k < 3; ++k)
#pragma unroll
            for (int r = 0; r < 8; ++r) acc[k][r] = 0.0f;
#pragma unroll
        for (int kc = 0; kc < 3; ++kc) {
            AU a;
            const f16* wr = &sW1[(mt * 16 + ln) * CPAD + kc * 32 + hl * 16];
            a.h[0] = *(const v8h*)(wr);        // K = k0 .. k0+7
            a.h[1] = *(const v8h*)(wr + 8);    // K = k0+8 .. k0+15
#pragma unroll
            for (int k = 0; k < 3; ++k)        // 3 independent WMMAs per A load
                acc[k] = __builtin_amdgcn_wmma_f32_16x16x32_f16(false, a.v, false, bx[k][kc],
                                                                (short)0, acc[k], false, false);
        }
        const int chb = mt * 16 + hl * 8;
        v8f iv = *(const v8f*)(&sInv1[chb]);
        v8f bt = *(const v8f*)(&sBeta1[chb]);
#pragma unroll
        for (int k = 0; k < 3; ++k) {
            v8h o;
#pragma unroll
            for (int r = 0; r < 8; ++r) {
                float y = acc[k][r] * iv[r] + bt[r];
                o[r] = (f16)(y > 0.0f ? y : 0.0f);
            }
            *(v8h*)(&sScr[((wv * 3 + k) * 16 + ln) * L1 + chb]) = o;  // stride 128 ch
        }
    }
    asm volatile("s_wait_dscnt 0" ::: "memory");

    // ---- layer 2: (64 x 128) @ (128 x 16) x3 ----
    v16h by[3][4];
#pragma unroll
    for (int k = 0; k < 3; ++k) {
        const f16* ycol = &sScr[((wv * 3 + k) * 16 + ln) * L1];
#pragma unroll
        for (int kc = 0; kc < 4; ++kc)
            by[k][kc] = *(const v16h*)(ycol + kc * 32 + hl * 16);
    }
    asm volatile("" ::: "memory");
#pragma unroll
    for (int mt = 0; mt < 4; ++mt) {
        v8f acc[3];
#pragma unroll
        for (int k = 0; k < 3; ++k)
#pragma unroll
            for (int r = 0; r < 8; ++r) acc[k][r] = 0.0f;
#pragma unroll
        for (int kc = 0; kc < 4; ++kc) {
            AU a;
            const f16* wr = &sW2[(mt * 16 + ln) * L1 + kc * 32 + hl * 16];
            a.h[0] = *(const v8h*)(wr);
            a.h[1] = *(const v8h*)(wr + 8);
#pragma unroll
            for (int k = 0; k < 3; ++k)
                acc[k] = __builtin_amdgcn_wmma_f32_16x16x32_f16(false, a.v, false, by[k][kc],
                                                                (short)0, acc[k], false, false);
        }
        const int chb = mt * 16 + hl * 8;
        v8f iv = *(const v8f*)(&sInv2[chb]);
        v8f bt = *(const v8f*)(&sBeta2[chb]);
#pragma unroll
        for (int k = 0; k < 3; ++k) {
            v8h o;
#pragma unroll
            for (int r = 0; r < 8; ++r) {
                float y = acc[k][r] * iv[r] + bt[r];
                o[r] = (f16)(y > 0.0f ? y : 0.0f);
            }
            *(v8h*)(&sScr[((wv * 3 + k) * 16 + ln) * L1 + ln * 0 + (0)] + ln * 0) ;
            *(v8h*)(&sScr[((wv * 3 + k) * 16) * L1] + ln * L2 + chb) = o;          // stride 64 ch
        }
    }
    asm volatile("s_wait_dscnt 0" ::: "memory");

    // ---- layer 3: (64 x 64) @ (64 x 16) x3, neighbor-sum in registers ----
    v16h bz[3][2];
#pragma unroll
    for (int k = 0; k < 3; ++k) {
        const f16* zcol = &sScr[((wv * 3 + k) * 16) * L1] + ln * L2;
#pragma unroll
        for (int kc = 0; kc < 2; ++kc)
            bz[k][kc] = *(const v16h*)(zcol + kc * 32 + hl * 16);
    }
    asm volatile("" ::: "memory");

    v8f outAcc[4];
#pragma unroll
    for (int mt = 0; mt < 4; ++mt) {
        v8f acc[3];
#pragma unroll
        for (int k = 0; k < 3; ++k)
#pragma unroll
            for (int r = 0; r < 8; ++r) acc[k][r] = 0.0f;
#pragma unroll
        for (int kc = 0; kc < 2; ++kc) {
            AU a;
            const f16* wr = &sW3[(mt * 16 + ln) * L2 + kc * 32 + hl * 16];
            a.h[0] = *(const v8h*)(wr);
            a.h[1] = *(const v8h*)(wr + 8);
#pragma unroll
            for (int k = 0; k < 3; ++k)
                acc[k] = __builtin_amdgcn_wmma_f32_16x16x32_f16(false, a.v, false, bz[k][kc],
                                                                (short)0, acc[k], false, false);
        }
        const int chb = mt * 16 + hl * 8;
        v8f iv = *(const v8f*)(&sInv3[chb]);
        v8f bt = *(const v8f*)(&sBeta3[chb]);
#pragma unroll
        for (int r = 0; r < 8; ++r) {
            float s = 0.0f;
#pragma unroll
            for (int k = 0; k < 3; ++k) {
                float y = acc[k][r] * iv[r] + bt[r];
                s += (y > 0.0f ? y : 0.0f);
            }
            outAcc[mt][r] = s;
        }
    }

    // ---- write (B,64,N): lanes 0..15 cover 16 consecutive n per store ----
    {
        int q = p0 + wv * 16 + ln;
        int bb = q >> 14, n = q & (NPTS - 1);
        float* op = out + (((size_t)bb * 64) << 14) + n;
#pragma unroll
        for (int mt = 0; mt < 4; ++mt)
#pragma unroll
            for (int r = 0; r < 8; ++r) {
                int ch = mt * 16 + hl * 8 + r;
                op[(size_t)ch << 14] = outAcc[mt][r];
            }
    }
}

extern "C" void kernel_launch(void* const* d_in, const int* in_sizes, int n_in,
                              void* d_out, int out_size, void* d_ws, size_t ws_size,
                              hipStream_t stream) {
    const float* src = (const float*)d_in[0];
    const float* tgt = (const float*)d_in[1];
    const float* fea = (const float*)d_in[2];
    const float* W1 = (const float*)d_in[3];
    const float* g1 = (const float*)d_in[4];
    const float* b1 = (const float*)d_in[5];
    const float* m1 = (const float*)d_in[6];
    const float* v1 = (const float*)d_in[7];
    const float* W2 = (const float*)d_in[8];
    const float* g2 = (const float*)d_in[9];
    const float* b2 = (const float*)d_in[10];
    const float* m2 = (const float*)d_in[11];
    const float* v2 = (const float*)d_in[12];
    const float* W3 = (const float*)d_in[13];
    const float* g3 = (const float*)d_in[14];
    const float* b3 = (const float*)d_in[15];
    const float* m3 = (const float*)d_in[16];
    const float* v3 = (const float*)d_in[17];
    float* out = (float*)d_out;

    dim3 grid((BATCH * NPTS) / PTS_PER_BLK);   // 2048 blocks
    fa_fused_wmma<<<grid, THREADS, 0, stream>>>(src, tgt, fea,
                                                W1, g1, b1, m1, v1,
                                                W2, g2, b2, m2, v2,
                                                W3, g3, b3, m3, v3,
                                                out);
}
